// LinearAttention_21414706937909
// MI455X (gfx1250) — matrix operand order repaired and checked by reading, not compiled, not run on hardware
//
#include <hip/hip_runtime.h>

// ---------------------------------------------------------------------------
// CDNA5 (gfx1250) chunked causal linear attention.
// bf16 WMMA w/ f32 accum; GEMMs staged through LDS by the Tensor Data Mover.
// ---------------------------------------------------------------------------

typedef __attribute__((ext_vector_type(16))) __bf16        v16bf;
typedef __attribute__((ext_vector_type(8)))  float         v8f;
typedef __attribute__((ext_vector_type(4)))  unsigned int  u32x4;
typedef __attribute__((ext_vector_type(8)))  unsigned int  u32x8;

union Frag {
    v16bf h;
    uint4 q[2];
    unsigned int u[8];
};

union Acc {
    v8f   v;
    float f[8];
};

__device__ __forceinline__ unsigned short f2bf(float f) {
    unsigned int u = __float_as_uint(f);
    u += 0x7FFFu + ((u >> 16) & 1u);   // round-to-nearest-even
    return (unsigned short)(u >> 16);
}

__device__ __forceinline__ float bf2f(unsigned short s) {
    return __uint_as_float(((unsigned int)s) << 16);
}

// A-fragment (16x32 bf16, MxK): lane<16 -> M=lane, K {0..7,16..23};
// lane>=16 -> M=lane-16, K {8..15,24..31}. Two 16-byte contiguous loads.
__device__ __forceinline__ Frag load_a(const unsigned short* src, int pitch,
                                       int row0, int kb) {
    int l = threadIdx.x & 31, lm = l & 15, hf = l >> 4;
    const unsigned short* p = src + (size_t)(row0 + lm) * pitch + kb + hf * 8;
    Frag f;
    f.q[0] = *(const uint4*)p;
    f.q[1] = *(const uint4*)(p + 16);
    return f;
}

// B-fragment (32x16 bf16, KxN), source stored row-major [N][K] ("W layout"):
// lane<16 -> N=lane, K {0..7,16..23}; lane>=16 -> N=lane-16, K {8..15,24..31}.
__device__ __forceinline__ Frag load_b(const unsigned short* src, int pitch,
                                       int n0, int kb) {
    int l = threadIdx.x & 31, lm = l & 15, hf = l >> 4;
    const unsigned short* p = src + (size_t)(n0 + lm) * pitch + kb + hf * 8;
    Frag f;
    f.q[0] = *(const uint4*)p;
    f.q[1] = *(const uint4*)(p + 16);
    return f;
}

__device__ __forceinline__ v8f wmma_bf16(const Frag& a, const Frag& b, v8f c) {
    return __builtin_amdgcn_wmma_f32_16x16x32_bf16(
        /*neg_a=*/false, a.h, /*neg_b=*/false, b.h,
        /*c_mod=*/(short)0, c, /*reuse_a=*/false, /*reuse_b=*/false);
}

// ---------------------------------------------------------------------------
// Tensor Data Mover: DMA a 2D tile (tile_d1 rows x tile_d0 elems of 2-byte
// data) from global memory into LDS (packed row-major, pitch = tile_d0).
// D# group0/group1 per CDNA5 ISA; 2-group form (2D tensor).
// Issued per-wave; caller must restrict to one wave and fence with
// s_wait_tensorcnt + workgroup barrier.
// ---------------------------------------------------------------------------
__device__ __forceinline__ void tdm_load_tile2d(unsigned lds_addr,
                                                const void* gptr,
                                                unsigned tensor_d0,
                                                unsigned tensor_d1,
                                                unsigned tile_d0,
                                                unsigned tile_d1,
                                                unsigned long long stride0) {
    unsigned long long ga = (unsigned long long)(size_t)gptr;
    u32x4 g0;
    g0[0] = 1u;                                            // count=1, user D#
    g0[1] = lds_addr;                                      // LDS byte address
    g0[2] = (unsigned)ga;                                  // global_addr[31:0]
    g0[3] = (unsigned)((ga >> 32) & 0x1FFFFFFull)          // global_addr[56:32]
          | (2u << 30);                                    // type = 2 (image)
    u32x8 g1;
    g1[0] = 1u << 16;                                      // data_size = 2 bytes
    g1[1] = (tensor_d0 & 0xFFFFu) << 16;                   // tensor_dim0[15:0]
    g1[2] = (tensor_d0 >> 16) | ((tensor_d1 & 0xFFFFu) << 16);
    g1[3] = (tensor_d1 >> 16) | (tile_d0 << 16);           // tile_dim0
    g1[4] = tile_d1 & 0xFFFFu;                             // tile_dim1 (dim2=0)
    g1[5] = (unsigned)stride0;                             // dim0_stride[31:0]
    g1[6] = (unsigned)((stride0 >> 32) & 0xFFFFull);       // dim0_stride[47:32]
    g1[7] = 0u;
    asm volatile("tensor_load_to_lds %0, %1" :: "s"(g0), "s"(g1) : "memory");
}

// ---------------------------------------------------------------------------
// Kernel: fp32 -> bf16 conversion
// ---------------------------------------------------------------------------
__global__ __launch_bounds__(256) void cvt_k(const float* __restrict__ src,
                                             unsigned short* __restrict__ dst,
                                             int n) {
    int i = blockIdx.x * 256 + threadIdx.x;
    int stride = gridDim.x * 256;
    for (; i < n; i += stride) dst[i] = f2bf(src[i]);
}

// ---------------------------------------------------------------------------
// Kernel: C = A @ W^T  (A:[M,K] bf16 row-major, W:[N,K] bf16 row-major)
// MODE 0: bf16 out; MODE 1: bf16 out with elu(x)+1; MODE 2: f32 out + bias
// block = 256 (8 waves); block tile 128(M) x 128(N); wave = 16-row strip.
// A/W k-tiles (128x32) staged into LDS by the TDM, double-buffered so the
// DMA of tile k+1 overlaps WMMA on tile k.
// ---------------------------------------------------------------------------
template <int MODE>
__global__ __launch_bounds__(256) void gemm_k(const unsigned short* __restrict__ A,
                                              const unsigned short* __restrict__ W,
                                              const float* __restrict__ bias,
                                              unsigned short* __restrict__ obf,
                                              float* __restrict__ of32,
                                              int M, int N, int K) {
    __shared__ unsigned short Abuf[2][128][32];
    __shared__ unsigned short Wbuf[2][128][32];

    int wave = threadIdx.x >> 5;
    int l = threadIdx.x & 31, lm = l & 15, hf = l >> 4;
    int mb = blockIdx.y * 128;
    int nb = blockIdx.x * 128;
    int m0 = mb + wave * 16;

    Acc acc[8];
#pragma unroll
    for (int t = 0; t < 8; t++)
#pragma unroll
        for (int r = 0; r < 8; r++) acc[t].f[r] = 0.f;

    const unsigned aoff[2] = { (unsigned)(size_t)&Abuf[0][0][0],
                               (unsigned)(size_t)&Abuf[1][0][0] };
    const unsigned woff[2] = { (unsigned)(size_t)&Wbuf[0][0][0],
                               (unsigned)(size_t)&Wbuf[1][0][0] };

    if (wave == 0) {  // prologue: DMA first k-tiles into buffer 0
        tdm_load_tile2d(aoff[0], A + (size_t)mb * K, K, 128, 32, 128, K);
        tdm_load_tile2d(woff[0], W + (size_t)nb * K, K, 128, 32, 128, K);
    }

    int p = 0;
    for (int kk = 0; kk < K; kk += 32, p ^= 1) {
        if (wave == 0) {
            if (kk + 32 < K) {
                tdm_load_tile2d(aoff[p ^ 1], A + (size_t)mb * K + kk + 32,
                                K, 128, 32, 128, K);
                tdm_load_tile2d(woff[p ^ 1], W + (size_t)nb * K + kk + 32,
                                K, 128, 32, 128, K);
                // 2 newest in flight; the 2 feeding this iteration are done
                __builtin_amdgcn_s_wait_tensorcnt(2);
            } else {
                __builtin_amdgcn_s_wait_tensorcnt(0);
            }
        }
        __syncthreads();

        Frag fa = load_a(&Abuf[p][0][0], 32, wave * 16, 0);
#pragma unroll
        for (int t = 0; t < 8; t++) {
            Frag fb = load_b(&Wbuf[p][0][0], 32, t * 16, 0);
            acc[t].v = wmma_bf16(fa, fb, acc[t].v);
        }
        __syncthreads();  // buffer p free for the DMA issued next iteration
    }

#pragma unroll
    for (int t = 0; t < 8; t++) {
        int col = nb + t * 16 + lm;
#pragma unroll
        for (int r = 0; r < 8; r++) {
            int row = m0 + r + hf * 8;
            float v = acc[t].f[r];
            if (MODE == 1) v = (v > 0.f) ? (v + 1.f) : __expf(v);  // elu(x)+1
            if (MODE == 2)
                of32[(size_t)row * N + col] = v + bias[col];
            else
                obf[(size_t)row * N + col] = f2bf(v);
        }
    }
}

// ---------------------------------------------------------------------------
// Kernel: per-(b,h) chunk KV state with exclusive prefix.
//   ST[e][d] = sum_c V[c][e] * phiK[c][d]   (stored transposed so that the
//   later phi_q @ S_prev WMMA reads K-contiguous B rows)
// grid = B*H (64), block = 256. Running ST kept in wave registers (2 tiles/wave).
// ---------------------------------------------------------------------------
__global__ __launch_bounds__(256) void chunk_kv_k(const unsigned short* __restrict__ phiK,
                                                  const unsigned short* __restrict__ Vb,
                                                  unsigned short* __restrict__ STp,
                                                  float* __restrict__ Zp) {
    __shared__ unsigned short KT[64][136];  // [d][c]
    __shared__ unsigned short VT[64][136];  // [e][c]
    __shared__ float Zsh[64];

    int tid = threadIdx.x;
    int wave = tid >> 5, l = tid & 31, lm = l & 15, hf = l >> 4;
    int bh = blockIdx.x, b = bh >> 4, h = bh & 15;

    if (tid < 64) Zsh[tid] = 0.f;

    int t0 = wave * 2, t1 = wave * 2 + 1;     // 4x4 tile grid of 64x64 state
    int tm = (t0 >> 2) * 16;                  // shared e-tile for both
    int tn0 = (t0 & 3) * 16, tn1 = (t1 & 3) * 16;
    Acc a0, a1;
#pragma unroll
    for (int r = 0; r < 8; r++) { a0.f[r] = 0.f; a1.f[r] = 0.f; }

    for (int n = 0; n < 16; n++) {
        __syncthreads();  // previous iteration fully consumed KT/VT
        size_t rb = ((size_t)b * 2048 + (size_t)n * 128) * 1024 + h * 64;
        for (int i = tid; i < 128 * 32; i += 256) {
            int c = i >> 5, dp = (i & 31) * 2;
            unsigned kk = *(const unsigned*)(phiK + rb + (size_t)c * 1024 + dp);
            KT[dp][c]     = (unsigned short)(kk & 0xFFFFu);
            KT[dp + 1][c] = (unsigned short)(kk >> 16);
            unsigned vv = *(const unsigned*)(Vb + rb + (size_t)c * 1024 + dp);
            VT[dp][c]     = (unsigned short)(vv & 0xFFFFu);
            VT[dp + 1][c] = (unsigned short)(vv >> 16);
        }
        __syncthreads();

        // exclusive prefix: write state BEFORE accumulating chunk n
        size_t sb = ((size_t)bh * 16 + n) * 64 * 64;
#pragma unroll
        for (int r = 0; r < 8; r++) {
            int e = tm + r + hf * 8;
            STp[sb + (size_t)e * 64 + tn0 + lm] = f2bf(a0.f[r]);
            STp[sb + (size_t)e * 64 + tn1 + lm] = f2bf(a1.f[r]);
        }
        if (tid < 64) {
            Zp[((size_t)bh * 16 + n) * 64 + tid] = Zsh[tid];
            float z = 0.f;
            for (int c = 0; c < 128; c++) z += bf2f(KT[tid][c]);
            Zsh[tid] += z;
        }

        // ST += VT @ KT^T  (contraction over c = 128)
        for (int kk = 0; kk < 128; kk += 32) {
            Frag fa  = load_a(&VT[0][0], 136, tm, kk);
            Frag fb0 = load_b(&KT[0][0], 136, tn0, kk);
            Frag fb1 = load_b(&KT[0][0], 136, tn1, kk);
            a0.v = wmma_bf16(fa, fb0, a0.v);
            a1.v = wmma_bf16(fa, fb1, a1.v);
        }
    }
}

// ---------------------------------------------------------------------------
// Kernel: per-(b,h,n) chunk attention.
//   A = tril(phiq @ phik^T); den = rowsum(A) + phiq.Z_prev + eps
//   y = (A @ V + phiq @ S_prev) / den
// grid = B*H*N (1024), block = 256 (8 waves, 16-row strips).
// ---------------------------------------------------------------------------
__global__ __launch_bounds__(256) void attn_k(const unsigned short* __restrict__ phiQ,
                                              const unsigned short* __restrict__ phiK,
                                              const unsigned short* __restrict__ Vb,
                                              const unsigned short* __restrict__ STp,
                                              const float* __restrict__ Zp,
                                              unsigned short* __restrict__ Y) {
    __shared__ unsigned short Ash[128][136];  // masked A (bf16), [c][s]
    __shared__ unsigned short VTs[64][136];   // V transposed, [e][c]
    __shared__ float den[128];

    int tid = threadIdx.x;
    int wave = tid >> 5, l = tid & 31, lm = l & 15, hf = l >> 4;
    int bhn = blockIdx.x;
    int n = bhn & 15, bh = bhn >> 4, h = bh & 15, b = bh >> 4;
    int rowc = b * 2048 + n * 128;  // global row of chunk start
    const unsigned short* Q  = phiQ + h * 64;
    const unsigned short* Kp = phiK + h * 64;
    int m0 = wave * 16;

    // ---- phase 1: A = phiq @ phik^T  (K = 64) ----
    Acc acc[8];
#pragma unroll
    for (int t = 0; t < 8; t++)
#pragma unroll
        for (int r = 0; r < 8; r++) acc[t].f[r] = 0.f;

    for (int kk = 0; kk < 64; kk += 32) {
        Frag fa = load_a(Q, 1024, rowc + m0, kk);
#pragma unroll
        for (int t = 0; t < 8; t++) {
            Frag fb = load_b(Kp, 1024, rowc + t * 16, kk);
            acc[t].v = wmma_bf16(fa, fb, acc[t].v);
        }
    }

    // mask (causal), row sums, spill masked A to LDS as bf16
#pragma unroll
    for (int r = 0; r < 8; r++) {
        int c = m0 + r + hf * 8;
        float rv = 0.f;
#pragma unroll
        for (int t = 0; t < 8; t++) {
            int s = t * 16 + lm;
            float v = acc[t].f[r];
            if (s > c) v = 0.f;
            rv += v;
            Ash[c][s] = f2bf(v);
        }
        rv += __shfl_xor(rv, 1, 16);
        rv += __shfl_xor(rv, 2, 16);
        rv += __shfl_xor(rv, 4, 16);
        rv += __shfl_xor(rv, 8, 16);
        if (lm == 0) den[c] = rv;
    }

    // stage V chunk transposed into LDS
    {
        size_t rb = (size_t)rowc * 1024 + h * 64;
        for (int i = tid; i < 128 * 32; i += 256) {
            int c = i >> 5, dp = (i & 31) * 2;
            unsigned vv = *(const unsigned*)(Vb + rb + (size_t)c * 1024 + dp);
            VTs[dp][c]     = (unsigned short)(vv & 0xFFFFu);
            VTs[dp + 1][c] = (unsigned short)(vv >> 16);
        }
    }
    __syncthreads();

    // den += phiq . Z_prev + eps
    if (tid < 128) {
        const float* zp = Zp + (size_t)bhn * 64;
        const unsigned short* qp = Q + (size_t)(rowc + tid) * 1024;
        float dv = 1e-6f;
        for (int d = 0; d < 64; d++) dv += bf2f(qp[d]) * zp[d];
        den[tid] += dv;
    }
    __syncthreads();

    // ---- phase 2: num = A @ V + phiq @ S_prev ----
    Acc y4[4];
#pragma unroll
    for (int t = 0; t < 4; t++)
#pragma unroll
        for (int r = 0; r < 8; r++) y4[t].f[r] = 0.f;

    for (int kk = 0; kk < 128; kk += 32) {        // A @ V over s
        Frag fa = load_a(&Ash[0][0], 136, m0, kk);
#pragma unroll
        for (int t = 0; t < 4; t++) {
            Frag fb = load_b(&VTs[0][0], 136, t * 16, kk);
            y4[t].v = wmma_bf16(fa, fb, y4[t].v);
        }
    }
    const unsigned short* ST = STp + (size_t)bhn * 4096;  // [e][d], pitch 64
    for (int kk = 0; kk < 64; kk += 32) {         // phiq @ S_prev over d
        Frag fa = load_a(Q, 1024, rowc + m0, kk);
#pragma unroll
        for (int t = 0; t < 4; t++) {
            Frag fb = load_b(ST, 64, t * 16, kk);
            y4[t].v = wmma_bf16(fa, fb, y4[t].v);
        }
    }

    // divide by den and store y (bf16) for the output projection
#pragma unroll
    for (int r = 0; r < 8; r++) {
        int cl = m0 + r + hf * 8;
        float dinv = 1.f / den[cl];
        size_t row = (size_t)(rowc + cl);
#pragma unroll
        for (int t = 0; t < 4; t++)
            Y[row * 1024 + h * 64 + t * 16 + lm] = f2bf(y4[t].f[r] * dinv);
    }
}

// ---------------------------------------------------------------------------
// Launch
// ---------------------------------------------------------------------------
extern "C" void kernel_launch(void* const* d_in, const int* in_sizes, int n_in,
                              void* d_out, int out_size, void* d_ws, size_t ws_size,
                              hipStream_t stream) {
    const float* x  = (const float*)d_in[0];
    const float* Wq = (const float*)d_in[1];
    const float* Wk = (const float*)d_in[2];
    const float* Wv = (const float*)d_in[3];
    const float* Wp = (const float*)d_in[4];
    const float* bp = (const float*)d_in[5];

    const int M = 8192;   // B*T
    const int E = 1024;

    char* w = (char*)d_ws;
    unsigned short* xb  = (unsigned short*)w; w += (size_t)M * E * 2;
    unsigned short* wqb = (unsigned short*)w; w += (size_t)E * E * 2;
    unsigned short* wkb = (unsigned short*)w; w += (size_t)E * E * 2;
    unsigned short* wvb = (unsigned short*)w; w += (size_t)E * E * 2;
    unsigned short* wpb = (unsigned short*)w; w += (size_t)E * E * 2;
    unsigned short* phq = (unsigned short*)w; w += (size_t)M * E * 2;
    unsigned short* phk = (unsigned short*)w; w += (size_t)M * E * 2;
    unsigned short* vbf = (unsigned short*)w; w += (size_t)M * E * 2;
    unsigned short* yb  = (unsigned short*)w; w += (size_t)M * E * 2;
    unsigned short* stp = (unsigned short*)w; w += (size_t)1024 * 4096 * 2;  // [b,h,n][e][d]
    float*          zp  = (float*)w;          w += (size_t)1024 * 64 * 4;    // [b,h,n][d]

    cvt_k<<<512, 256, 0, stream>>>(x,  xb,  M * E);
    cvt_k<<<256, 256, 0, stream>>>(Wq, wqb, E * E);
    cvt_k<<<256, 256, 0, stream>>>(Wk, wkb, E * E);
    cvt_k<<<256, 256, 0, stream>>>(Wv, wvb, E * E);
    cvt_k<<<256, 256, 0, stream>>>(Wp, wpb, E * E);

    dim3 gg(E / 128, M / 128);  // (8, 64)
    gemm_k<1><<<gg, 256, 0, stream>>>(xb, wqb, nullptr, phq, nullptr, M, E, E);
    gemm_k<1><<<gg, 256, 0, stream>>>(xb, wkb, nullptr, phk, nullptr, M, E, E);
    gemm_k<0><<<gg, 256, 0, stream>>>(xb, wvb, nullptr, vbf, nullptr, M, E, E);

    chunk_kv_k<<<64, 256, 0, stream>>>(phk, vbf, stp, zp);
    attn_k<<<1024, 256, 0, stream>>>(phq, phk, vbf, stp, zp, yb);

    gemm_k<2><<<gg, 256, 0, stream>>>(yb, wpb, bp, nullptr, (float*)d_out, M, E, E);
}
